// LeWinAttention_884763263595
// MI455X (gfx1250) — matrix operand order repaired and checked by reading, not compiled, not run on hardware
//
#include <hip/hip_runtime.h>
#include <hip/hip_bf16.h>

// CDNA5 / gfx1250 WMMA + TDM types
typedef __attribute__((ext_vector_type(16))) __bf16       v16bf;
typedef __attribute__((ext_vector_type(8)))  float        v8f;
typedef __attribute__((ext_vector_type(4)))  unsigned int v4u;
typedef __attribute__((ext_vector_type(8)))  int          v8i;
typedef __attribute__((ext_vector_type(4)))  int          v4i;

#define HWPIX 16384   // H*W
#define CCH   128
#define NWIN  1024
#define QROWS 256
#define KROWS 512
#define HDIM  64
#define ATT_SCALE 0.125f   // 64^-0.5

#if __has_builtin(__builtin_amdgcn_tensor_load_to_lds)
#define USE_TDM 1
#else
#define USE_TDM 0
#endif

// ---------------- fragment helpers (wave32, per CDNA5 ISA) --------

__device__ __forceinline__ int lane_id() { return threadIdx.x & 31; }

__device__ __forceinline__ v8f zero8() {
  v8f z;
#pragma unroll
  for (int i = 0; i < 8; ++i) z[i] = 0.0f;
  return z;
}

__device__ __forceinline__ v8f wmma_bf16(v16bf a, v16bf b, v8f c) {
  return __builtin_amdgcn_wmma_f32_16x16x32_bf16(false, a, false, b, (short)0, c,
                                                 false, false);
}

// A-matrix 16x32 bf16 from row-major source: two contiguous 16B runs per lane.
// lanes 0-15: M=lane, K={0..7,16..23}; lanes 16-31: M=lane-16, K={8..15,24..31}
__device__ __forceinline__ v16bf load_a_bf16(const __bf16* base, int ld) {
  int lane = lane_id();
  int m  = lane & 15;
  int hk = (lane >> 4) << 3;   // 0 or 8
  const __bf16* row = base + (size_t)m * ld;
  union { uint4 u[2]; v16bf v; } t;
  t.u[0] = *(const uint4*)(row + hk);
  t.u[1] = *(const uint4*)(row + 16 + hk);
  return t.v;
}

// Same A layout gathering from f32 weights (compiler vectorizes to b128+cvt).
__device__ __forceinline__ v16bf load_a_f32(const float* base, int ld) {
  int lane = lane_id();
  int m  = lane & 15;
  int hk = (lane >> 4) << 3;
  const float* row = base + (size_t)m * ld;
  v16bf a;
#pragma unroll
  for (int v = 0; v < 4; ++v) {
    a[2*v]   = (__bf16)row[hk + 2*v];
    a[2*v+1] = (__bf16)row[hk + 2*v + 1];
    a[2*v+8] = (__bf16)row[16 + hk + 2*v];
    a[2*v+9] = (__bf16)row[16 + hk + 2*v + 1];
  }
  return a;
}

// B-matrix 32x16 bf16, N-major source: B[k][n] = base[n*ld + k].
// K={0..7,16..23} (lanes 0-15) / {8..15,24..31} (lanes 16-31) -> two 16B loads.
__device__ __forceinline__ v16bf load_b_n(const __bf16* base, int ld) {
  int lane = lane_id();
  int n    = lane & 15;
  int koff = (lane >> 4) << 3;   // 0 or 8
  const __bf16* row = base + (size_t)n * ld + koff;
  union { uint4 u[2]; v16bf v; } t;
  t.u[0] = *(const uint4*)(row);
  t.u[1] = *(const uint4*)(row + 16);
  return t.v;
}

__device__ __forceinline__ float rowmax16(float x) {
#pragma unroll
  for (int off = 8; off; off >>= 1) x = fmaxf(x, __shfl_xor(x, off, 32));
  return x;
}
__device__ __forceinline__ float rowsum16(float x) {
#pragma unroll
  for (int off = 8; off; off >>= 1) x += __shfl_xor(x, off, 32);
  return x;
}

// ---------------- Tensor Data Mover: 2D tile -> LDS (rows padded to 72) -----
// D# per CDNA5 ISA. data_size=2B; pad_interval=4 (32 DWORDs
// = one 64-elem bf16 row), pad_amount=3 (4 DWORDs = 8 bf16) -> LDS stride 72.
// 6-arg builtin form (clang-23 / therock-10.0).

__device__ __forceinline__ void tdm_load_2d(unsigned lds_addr, const void* gptr,
                                            unsigned tile_d0, unsigned tile_d1,
                                            unsigned long long d0_stride,
                                            unsigned tensor_d0, unsigned tensor_d1)
{
#if USE_TDM
  unsigned long long ga = (unsigned long long)gptr;
  v4u g0;
  g0[0] = 1u;                                          // count=1 (valid user D#)
  g0[1] = lds_addr;                                    // LDS byte address
  g0[2] = (unsigned)(ga & 0xffffffffu);                // global_addr[31:0]
  g0[3] = (unsigned)((ga >> 32) & 0x01ffffffu) | (2u << 30);  // addr[56:32]|type=2
  unsigned w0 = (1u << 16)    // data_size = 2 bytes
              | (1u << 20)    // pad_enable
              | (4u << 22)    // pad_interval = 32 DWORDs
              | (3u << 25);   // pad_amount  = 4 DWORDs
  v8i g1;
  g1[0] = (int)w0;
  g1[1] = (int)((tensor_d0 & 0xffffu) << 16);                           // dim0 lo
  g1[2] = (int)(((tensor_d0 >> 16) & 0xffffu) | ((tensor_d1 & 0xffffu) << 16));
  g1[3] = (int)(((tensor_d1 >> 16) & 0xffffu) | ((tile_d0 & 0xffffu) << 16));
  g1[4] = (int)(tile_d1 & 0xffffu);                                     // tile_dim1
  g1[5] = (int)(unsigned)(d0_stride & 0xffffffffu);                     // stride lo
  g1[6] = (int)(unsigned)((d0_stride >> 32) & 0xffffu);                 // stride hi
  g1[7] = 0;
  v4i z4 = {0, 0, 0, 0};
  v8i z8 = {0, 0, 0, 0, 0, 0, 0, 0};
  __builtin_amdgcn_tensor_load_to_lds(g0, g1, z4, z4, z8, 0);
#endif
}

// ---------------- kernel 1: fused QKV projection + RoPE, f32 -> bf16 --------

__global__ __launch_bounds__(256) void proj_qkv_rope(
    const float* __restrict__ x, const float* __restrict__ skip,
    const float* __restrict__ sinb, const float* __restrict__ cosb,
    const float* __restrict__ Wq, const float* __restrict__ Wkv,
    __bf16* __restrict__ q_ws, __bf16* __restrict__ k_ws,
    __bf16* __restrict__ vT_ws)
{
  __shared__ __bf16 sXT[64][136];  // transposed x tile: 64 pixels x 128 channels

  int pt = blockIdx.x;             // 256 pixel tiles of 64
  int bi = blockIdx.y;             // 0..3 kv batches; 0..1 also Q
  int pbase = pt * 64;
  const float* src = (bi < 2) ? (x + (size_t)bi * CCH * HWPIX)
                              : (skip + (size_t)(bi - 2) * CCH * HWPIX);

  for (int i = threadIdx.x; i < CCH * 64; i += 256) {
    int c = i >> 6, p = i & 63;    // coalesced global read, transposed LDS write
    sXT[p][c] = (__bf16)src[(size_t)c * HWPIX + pbase + p];
  }
  __syncthreads();

  int wave  = threadIdx.x >> 5;
  int lane  = threadIdx.x & 31;
  int ptile = wave & 3;
  int owav  = wave >> 2;
  int n  = lane & 15;
  int mo = (lane >> 4) << 3;
  int p  = pbase + ptile * 16 + n;
  int h  = p >> 7, w = p & 127;
  int d  = w & 63;

  int NOT = (bi < 2) ? 96 : 64;
  for (int ot = owav; ot < NOT; ot += 2) {
    bool isq = (bi < 2) && (ot < 32);
    int r0 = isq ? ot * 16 : ((bi < 2) ? (ot - 32) * 16 : ot * 16);
    const float* wbase = isq ? (Wq + (size_t)r0 * CCH) : (Wkv + (size_t)r0 * CCH);

    v8f acc = zero8();
#pragma unroll
    for (int kc = 0; kc < 4; ++kc) {
      v16bf a = load_a_f32(wbase + kc * 32, CCH);
      v16bf b = load_b_n(&sXT[ptile * 16][kc * 32], 136);
      acc = wmma_bf16(a, b, acc);
    }

    int mode = isq ? 0 : (r0 < 512 ? 1 : 2);
#pragma unroll
    for (int v = 0; v < 8; ++v) {
      float val = acc[v];
      int o = (mode == 2 ? r0 - 512 : r0) + mo + v;
      if (mode != 2 && d < 32) {
        int h2 = (2 * o + (h >> 6)) & 127;
        int w2 = (2 * h + (w >> 6)) & 127;
        size_t si = ((size_t)h2 * 128 + w2) * 32 + d;
        float c = cosb[si], s = sinb[si];
        float sg = (d & 1) ? 1.0f : -1.0f;
        val = (mode == 0) ? (val * c + sg * val * s)   // q: q*cos + rot(q)*sin
                          : (val + c + sg * val * s);  // k (reference quirk)
      }
      if (mode == 0) {
        q_ws[((size_t)bi * 512 + o) * HWPIX + p] = (__bf16)val;
      } else if (mode == 1) {
        k_ws[((size_t)bi * 512 + o) * HWPIX + p] = (__bf16)val;
      } else {
        size_t flat = ((size_t)bi * 512 + o) * HWPIX + p;   // window reinterpret
        size_t win = flat >> 15, key = (flat >> 6) & 511, dd = flat & 63;
        vT_ws[(win << 15) + (dd << 9) + key] = (__bf16)val;
      }
    }
  }
}

// ---------------- kernel 2: windowed flash attention, TDM double-buffered ---

__global__ __launch_bounds__(256) void win_attn(
    const __bf16* __restrict__ q_ws, const __bf16* __restrict__ k_ws,
    const __bf16* __restrict__ vT_ws, __bf16* __restrict__ oT_ws)
{
  __shared__ __bf16 sQ[QROWS][72];
  __shared__ __bf16 sK[2][64][72];    // double-buffered: 64 keys x 64 d
  __shared__ __bf16 sVT[2][64][72];   // double-buffered: 64 d x 64 keys
  __shared__ __bf16 sP[8][16][40];

  int win = blockIdx.x;
  const __bf16* qsrc  = q_ws  + (size_t)win * QROWS * HDIM;
  const __bf16* ksrc  = k_ws  + (size_t)win * KROWS * HDIM;
  const __bf16* vTsrc = vT_ws + ((size_t)win << 15);   // (64 d) x (512 keys)

  int wave = threadIdx.x >> 5;
  int lane = threadIdx.x & 31;
  int n  = lane & 15;
  int mo = (lane >> 4) << 3;

#if USE_TDM
  if (wave == 0) {
    // prologue: Q window + first K/V chunk; TDM ops retire in order
    tdm_load_2d((unsigned)(size_t)&sQ[0][0], qsrc, 64, QROWS, 64, 64, QROWS);
    tdm_load_2d((unsigned)(size_t)&sK[0][0][0], ksrc, 64, 64, 64, 64, 64);
    tdm_load_2d((unsigned)(size_t)&sVT[0][0][0], vTsrc, 64, 64, 512, 512, 64);
    __builtin_amdgcn_s_wait_tensorcnt(0);
  }
#else
  for (int i = threadIdx.x; i < QROWS * 4; i += 256) {
    int r = i >> 2, part = i & 3;
    const uint4* s = (const uint4*)(qsrc + (size_t)r * HDIM + part * 16);
    uint4* dq = (uint4*)(&sQ[r][part * 16]);
    dq[0] = s[0]; dq[1] = s[1];
  }
#endif
  __syncthreads();

  v16bf aq[2][2];
#pragma unroll
  for (int m = 0; m < 2; ++m)
#pragma unroll
    for (int c = 0; c < 2; ++c)
      aq[m][c] = load_a_bf16(&sQ[wave * 32 + m * 16][c * 32], 72);

  v8f   oacc[2][4];
  float mrun[2][8], lrun[2][8];
#pragma unroll
  for (int m = 0; m < 2; ++m) {
#pragma unroll
    for (int dt = 0; dt < 4; ++dt) oacc[m][dt] = zero8();
#pragma unroll
    for (int v = 0; v < 8; ++v) { mrun[m][v] = -1e30f; lrun[m][v] = 0.0f; }
  }

  for (int ch = 0; ch < 8; ++ch) {     // 512 keys in chunks of 64
    int buf = ch & 1;
#if USE_TDM
    if (wave == 0) {
      if (ch < 7) {
        // stream next chunk into the other buffer while this one computes
        int nb = (ch + 1) & 1;
        tdm_load_2d((unsigned)(size_t)&sK[nb][0][0],
                    ksrc + (size_t)(ch + 1) * 64 * HDIM, 64, 64, 64, 64, 64);
        tdm_load_2d((unsigned)(size_t)&sVT[nb][0][0],
                    vTsrc + (size_t)(ch + 1) * 64, 64, 64, 512, 512, 64);
        __builtin_amdgcn_s_wait_tensorcnt(2);   // current chunk complete
      } else {
        __builtin_amdgcn_s_wait_tensorcnt(0);
      }
    }
#else
    for (int i = threadIdx.x; i < 64 * 4; i += 256) {
      int r = i >> 2, part = i & 3;
      const uint4* sk = (const uint4*)(ksrc + (size_t)(ch * 64 + r) * HDIM + part * 16);
      uint4* dk = (uint4*)(&sK[buf][r][part * 16]);
      dk[0] = sk[0]; dk[1] = sk[1];
      const uint4* sv = (const uint4*)(vTsrc + (size_t)r * KROWS + ch * 64 + part * 16);
      uint4* dv = (uint4*)(&sVT[buf][r][part * 16]);
      dv[0] = sv[0]; dv[1] = sv[1];
    }
#endif
    if (ch < 7 && threadIdx.x < 64) {  // global_prefetch_b8 for next chunk
      __builtin_prefetch(ksrc + (size_t)((ch + 1) * 64 + threadIdx.x) * HDIM, 0, 1);
      __builtin_prefetch(vTsrc + (size_t)threadIdx.x * KROWS + (ch + 1) * 64, 0, 1);
    }
    __syncthreads();

#pragma unroll
    for (int sub = 0; sub < 2; ++sub) {
      int kb = sub * 32;
#pragma unroll
      for (int m = 0; m < 2; ++m) {
        v8f s0 = zero8(), s1 = zero8();
        {
          v16bf b00 = load_b_n(&sK[buf][kb][0],       72);
          v16bf b01 = load_b_n(&sK[buf][kb][32],      72);
          s0 = wmma_bf16(aq[m][0], b00, s0);
          s0 = wmma_bf16(aq[m][1], b01, s0);
          v16bf b10 = load_b_n(&sK[buf][kb + 16][0],  72);
          v16bf b11 = load_b_n(&sK[buf][kb + 16][32], 72);
          s1 = wmma_bf16(aq[m][0], b10, s1);
          s1 = wmma_bf16(aq[m][1], b11, s1);
        }

        float mnew[8], al[8];
#pragma unroll
        for (int v = 0; v < 8; ++v) {
          float x0 = s0[v] * ATT_SCALE, x1 = s1[v] * ATT_SCALE;
          s0[v] = x0; s1[v] = x1;
          float mx = rowmax16(fmaxf(x0, x1));
          mnew[v] = fmaxf(mrun[m][v], mx);
          al[v]   = __expf(mrun[m][v] - mnew[v]);
          mrun[m][v] = mnew[v];
        }
#pragma unroll
        for (int v = 0; v < 8; ++v) {
          float p0 = __expf(s0[v] - mnew[v]);
          float p1 = __expf(s1[v] - mnew[v]);
          sP[wave][mo + v][n]      = (__bf16)p0;
          sP[wave][mo + v][16 + n] = (__bf16)p1;
          lrun[m][v] = lrun[m][v] * al[v] + rowsum16(p0 + p1);
        }
#pragma unroll
        for (int dt = 0; dt < 4; ++dt)
#pragma unroll
          for (int v = 0; v < 8; ++v) oacc[m][dt][v] *= al[v];

        v16bf pf = load_a_bf16(&sP[wave][0][0], 40);
#pragma unroll
        for (int dt = 0; dt < 4; ++dt) {
          // B[k=key][n=d] = sVT[d][key]: N-major contiguous
          v16bf bv = load_b_n(&sVT[buf][dt * 16][kb], 72);
          oacc[m][dt] = wmma_bf16(pf, bv, oacc[m][dt]);
        }
      }
    }
    __syncthreads();   // all waves done with buf before TDM refills it
  }

  // normalize; store pixel-major (b, p, o) so kernel 3 B-frags are contiguous
#pragma unroll
  for (int m = 0; m < 2; ++m)
#pragma unroll
    for (int dt = 0; dt < 4; ++dt)
#pragma unroll
      for (int v = 0; v < 8; ++v) {
        int row = wave * 32 + m * 16 + mo + v;
        int col = dt * 16 + n;
        size_t flat = (((size_t)win * QROWS + row) << 6) + col;  // (b,o,h,w) flat
        size_t b = flat >> 23, o = (flat >> 14) & 511, p = flat & 16383;
        oT_ws[(b << 23) + p * 512 + o] =
            (__bf16)(oacc[m][dt][v] / lrun[m][v]);
      }
}

// ---------------- kernel 3: output projection (bf16 -> f32) -----------------

__global__ __launch_bounds__(256) void proj_out(
    const __bf16* __restrict__ attT, const float* __restrict__ Wout,
    float* __restrict__ out)
{
  int pt = blockIdx.x;
  int b  = blockIdx.y;
  int pbase = pt * 64;
  int wave  = threadIdx.x >> 5;
  int lane  = threadIdx.x & 31;
  int ptile = wave & 3;
  int owav  = wave >> 2;
  int n  = lane & 15;
  int mo = (lane >> 4) << 3;
  int p0 = pbase + ptile * 16;
  const __bf16* abase = attT + ((size_t)b << 23);   // (p, o) pixel-major

  for (int ot = owav; ot < 8; ot += 2) {
    v8f acc = zero8();
#pragma unroll
    for (int kc = 0; kc < 16; ++kc) {
      v16bf a  = load_a_f32(Wout + (size_t)(ot * 16) * 512 + kc * 32, 512);
      v16bf bf = load_b_n(abase + (size_t)p0 * 512 + kc * 32, 512);
      acc = wmma_bf16(a, bf, acc);
    }
    float* dst = out + ((size_t)b * 128 + ot * 16) * HWPIX + p0;
#pragma unroll
    for (int v = 0; v < 8; ++v)
      dst[(size_t)(mo + v) * HWPIX + n] = acc[v];
  }
}

// ---------------- launch ----------------------------------------------------

extern "C" void kernel_launch(void* const* d_in, const int* in_sizes, int n_in,
                              void* d_out, int out_size, void* d_ws, size_t ws_size,
                              hipStream_t stream)
{
  const float* x    = (const float*)d_in[0];
  const float* skip = (const float*)d_in[1];
  const float* sinb = (const float*)d_in[2];
  const float* cosb = (const float*)d_in[3];
  const float* Wq   = (const float*)d_in[4];
  const float* Wkv  = (const float*)d_in[5];
  const float* Wout = (const float*)d_in[6];
  float* out = (float*)d_out;

  // ws: bf16 q (33.5MB) | k (67MB) | vT (67MB) | attn-out transposed (33.5MB)
  __bf16* q_ws  = (__bf16*)d_ws;
  __bf16* k_ws  = q_ws  + (size_t)2 * 512 * HWPIX;
  __bf16* vT_ws = k_ws  + (size_t)4 * 512 * HWPIX;
  __bf16* oT_ws = vT_ws + (size_t)4 * 512 * HWPIX;

  proj_qkv_rope<<<dim3(256, 4), 256, 0, stream>>>(x, skip, sinb, cosb, Wq, Wkv,
                                                  q_ws, k_ws, vT_ws);
  win_attn<<<dim3(NWIN), 256, 0, stream>>>(q_ws, k_ws, vT_ws, oT_ws);
  proj_out<<<dim3(256, 2), 256, 0, stream>>>(oT_ws, Wout, out);
}
